// SelfAttentionBlock_75814762709507
// MI455X (gfx1250) — matrix operand order repaired and checked by reading, not compiled, not run on hardware
//
#include <hip/hip_runtime.h>
#include <hip/hip_bf16.h>

// CDNA5 WMMA vector types
typedef __attribute__((ext_vector_type(16))) __bf16 v16bf;
typedef __attribute__((ext_vector_type(4)))  __bf16 v4bf;
typedef __attribute__((ext_vector_type(8)))  float  v8f;

// Exact parameter type of the async-copy builtin (per hipcc diagnostic):
// int __attribute__((vector_size(16))) in AS1 (global) / AS3 (LDS).
typedef int v4i_vs __attribute__((vector_size(16)));
typedef v4i_vs __attribute__((address_space(1))) as1_v4i;
typedef v4i_vs __attribute__((address_space(3))) as3_v4i;

#define WMMA_BF16(a, b, c) \
  __builtin_amdgcn_wmma_f32_16x16x32_bf16(false, (a), false, (b), (short)0, (c), false, false)

// gfx1250 async global->LDS copy path (ASYNCcnt-tracked), if the toolchain has it.
#if defined(__has_builtin)
#if __has_builtin(__builtin_amdgcn_global_load_async_to_lds_b128)
#define SAB_ASYNC_LDS 1
#endif
#endif

// ---------------------------------------------------------------------------
// Kernel 1: 1x1-conv projection as GEMM  Y[b,o,p] = sum_c W[o,c] * X[b,c,p] + bias[o]
//   X: fp32 [B=16][C=512][P=4096], W: fp32 [512][512], Y: bf16 [16][512][4096]
//   Grid: (P/64, O/64, B) = (64, 8, 16), Block: 128 (4 waves).
//   A tile staged [o][k] (row-contiguous A-frag reads -> ds_load_2addr_b64),
//   B tile staged TRANSPOSED [p][k] so each lane's 16 K-values are contiguous.
// ---------------------------------------------------------------------------
__global__ __launch_bounds__(128) void sab_proj_bf16_wmma(
    const float* __restrict__ X, const float* __restrict__ Wm,
    const float* __restrict__ bias, __bf16* __restrict__ Y)
{
  constexpr int C = 512, P = 4096, O = 512;
  const int b    = blockIdx.z;
  const int o0   = blockIdx.y * 64;
  const int p0   = blockIdx.x * 64;
  const int t    = threadIdx.x;
  const int wid  = t >> 5;
  const int lane = t & 31;
  const int mrow = lane & 15;
  const int kA   = (lane < 16) ? 0 : 16;  // A-fragment K pattern (16-bit 16x32)
  const int kB   = (lane < 16) ? 0 : 16;  // B-fragment K pattern (32x16)

  __shared__ __bf16 As[64][36];   // [o][k], padded
  __shared__ __bf16 BsT[64][36];  // [p][k] (transposed), padded

  const float* __restrict__ Xb = X + (size_t)b * C * P;

  v8f acc0 = {}, acc1 = {}, acc2 = {}, acc3 = {};

  for (int kk = 0; kk < C; kk += 32) {
    // Stage A tile: W[o0..o0+63][kk..kk+31], fp32 -> bf16, row-major
#pragma unroll
    for (int i = 0; i < 4; ++i) {
      const int flat = (t + i * 128) * 4;          // 2048 elems / 128 thr
      const int r = flat >> 5, c = flat & 31;
      const float4 f = *reinterpret_cast<const float4*>(
          &Wm[(size_t)(o0 + r) * C + (kk + c)]);
      v4bf h;
      h[0] = (__bf16)f.x; h[1] = (__bf16)f.y; h[2] = (__bf16)f.z; h[3] = (__bf16)f.w;
      *reinterpret_cast<v4bf*>(&As[r][c]) = h;
    }
    // Stage B tile transposed: BsT[p][k] = X[b][kk+k][p0+p].
    // Gather 4 K-values per thread; adjacent lanes touch adjacent p -> coalesced.
#pragma unroll
    for (int i = 0; i < 4; ++i) {
      const int chunk = t + i * 128;               // 512 chunks of 4 elems
      const int p = chunk & 63;
      const int k = (chunk >> 6) * 4;
      const float* __restrict__ xp = &Xb[(size_t)(kk + k) * P + (p0 + p)];
      v4bf h;
      h[0] = (__bf16)xp[0];
      h[1] = (__bf16)xp[P];
      h[2] = (__bf16)xp[2 * (size_t)P];
      h[3] = (__bf16)xp[3 * (size_t)P];
      *reinterpret_cast<v4bf*>(&BsT[p][k]) = h;
    }
    if (kk + 32 < C) {  // hint next X K-tile toward the caches
      __builtin_prefetch(&Xb[(size_t)(kk + 32) * P + p0 + (t & 63)], 0, 1);
    }
    __syncthreads();

    v16bf a;
#pragma unroll
    for (int i = 0; i < 8; ++i) {
      a[i]     = As[wid * 16 + mrow][kA + i];
      a[i + 8] = As[wid * 16 + mrow][kA + 8 + i];
    }
    v16bf b0, b1, b2, b3;  // contiguous 16 bf16 per lane
#pragma unroll
    for (int i = 0; i < 16; ++i) {
      b0[i] = BsT[ 0 + mrow][kB + i];
      b1[i] = BsT[16 + mrow][kB + i];
      b2[i] = BsT[32 + mrow][kB + i];
      b3[i] = BsT[48 + mrow][kB + i];
    }
    acc0 = WMMA_BF16(a, b0, acc0);
    acc1 = WMMA_BF16(a, b1, acc1);
    acc2 = WMMA_BF16(a, b2, acc2);
    acc3 = WMMA_BF16(a, b3, acc3);
    __syncthreads();
  }

  const int rb = (lane < 16) ? 0 : 8;  // f32 C/D layout: row = r + 8*(lane>=16)
#pragma unroll
  for (int r = 0; r < 8; ++r) {
    const int orow = o0 + wid * 16 + rb + r;
    const float bv = bias[orow];
    __bf16* yp = &Y[((size_t)b * O + orow) * P + p0 + mrow];
    yp[ 0] = (__bf16)(acc0[r] + bv);
    yp[16] = (__bf16)(acc1[r] + bv);
    yp[32] = (__bf16)(acc2[r] + bv);
    yp[48] = (__bf16)(acc3[r] + bv);
  }
}

// ---------------------------------------------------------------------------
// Kernel 2: scores[b,o,w,x] = (1/sqrt(C)) * sum_h q[b,o,w,h] * k[b,o,x,h]
//   One workgroup (4 waves) per (b,o). 64x64 output, K=64 in two steps of 32.
//   q/k tiles are pure bf16 copies -> stage via GLOBAL_LOAD_ASYNC_TO_LDS_B128
//   (ASYNCcnt) when available; B = k^T so kt's [x][h] layout gives contiguous
//   per-lane K reads. Tiles padded to 72 cols (144B rows) for 16B B128 alignment.
// ---------------------------------------------------------------------------
__global__ __launch_bounds__(128) void sab_scores_wmma(
    const __bf16* __restrict__ Q, const __bf16* __restrict__ Kp,
    float* __restrict__ S, float rscale)
{
  constexpr int P = 4096, O = 512;
  const int o = blockIdx.x, b = blockIdx.y;
  const int t = threadIdx.x, wid = t >> 5, lane = t & 31;
  const int mrow = lane & 15;
  const int kA = (lane < 16) ? 0 : 16;
  const int kB = (lane < 16) ? 0 : 16;

  __shared__ __bf16 qt[64][72];  // [w][h], 144B rows (16B-aligned, conflict-free)
  __shared__ __bf16 kt[64][72];  // [x][h]

  const size_t base = ((size_t)b * O + o) * P;
#ifdef SAB_ASYNC_LDS
#pragma unroll
  for (int i = 0; i < 4; ++i) {
    const int e = (t + i * 128) * 8;             // 512 chunks of 8 bf16 (16B)
    const int r = e >> 6, c = e & 63;
    __builtin_amdgcn_global_load_async_to_lds_b128(
        (as1_v4i*)(unsigned long long)(const void*)&Q[base + e],
        (as3_v4i*)(unsigned)(unsigned long long)(void*)&qt[r][c],
        0, 0);
    __builtin_amdgcn_global_load_async_to_lds_b128(
        (as1_v4i*)(unsigned long long)(const void*)&Kp[base + e],
        (as3_v4i*)(unsigned)(unsigned long long)(void*)&kt[r][c],
        0, 0);
  }
#if __has_builtin(__builtin_amdgcn_s_wait_asynccnt)
  __builtin_amdgcn_s_wait_asynccnt(0);
#else
  asm volatile("s_wait_asynccnt 0x0" ::: "memory");
#endif
#else
#pragma unroll
  for (int i = 0; i < 8; ++i) {
    const int flat = (t + i * 128) * 4;          // 4096 bf16 / 128 thr
    const int r = flat >> 6, c = flat & 63;
    *reinterpret_cast<v4bf*>(&qt[r][c]) = *reinterpret_cast<const v4bf*>(&Q[base + flat]);
    *reinterpret_cast<v4bf*>(&kt[r][c]) = *reinterpret_cast<const v4bf*>(&Kp[base + flat]);
  }
#endif
  __syncthreads();

  v8f acc0 = {}, acc1 = {}, acc2 = {}, acc3 = {};
  const int w0 = wid * 16;
#pragma unroll
  for (int kk = 0; kk < 64; kk += 32) {
    v16bf a;
#pragma unroll
    for (int i = 0; i < 8; ++i) {
      a[i]     = qt[w0 + mrow][kk + kA + i];
      a[i + 8] = qt[w0 + mrow][kk + kA + 8 + i];
    }
    v16bf b0, b1, b2, b3;  // B[krel][x] = kt[x][kk+krel] (row-contiguous read)
#pragma unroll
    for (int i = 0; i < 16; ++i) {
      b0[i] = kt[ 0 + mrow][kk + kB + i];
      b1[i] = kt[16 + mrow][kk + kB + i];
      b2[i] = kt[32 + mrow][kk + kB + i];
      b3[i] = kt[48 + mrow][kk + kB + i];
    }
    acc0 = WMMA_BF16(a, b0, acc0);
    acc1 = WMMA_BF16(a, b1, acc1);
    acc2 = WMMA_BF16(a, b2, acc2);
    acc3 = WMMA_BF16(a, b3, acc3);
  }

  const int rb = (lane < 16) ? 0 : 8;
#pragma unroll
  for (int r = 0; r < 8; ++r) {
    float* sp = &S[base + (size_t)(w0 + rb + r) * 64 + mrow];
    sp[ 0] = acc0[r] * rscale;
    sp[16] = acc1[r] * rscale;
    sp[32] = acc2[r] * rscale;
    sp[48] = acc3[r] * rscale;
  }
}

// ---------------------------------------------------------------------------
// Kernel 3: channel-softmax stats. softmax is over O (dim=1): for each (b,w,x)
// reduce max and sum(exp) over the 512 channels. S mostly L2-resident (134MB<192MB).
// ---------------------------------------------------------------------------
__global__ __launch_bounds__(256) void sab_softmax_stats(
    const float* __restrict__ S, float* __restrict__ M, float* __restrict__ Zr)
{
  const int idx = blockIdx.x * 256 + threadIdx.x;  // 16 * 4096 threads
  const int b  = idx >> 12;
  const int wx = idx & 4095;
  const float* __restrict__ sp = S + (size_t)b * 512 * 4096 + wx;
  float m = -3.402823466e38f;
  for (int o = 0; o < 512; ++o) m = fmaxf(m, sp[(size_t)o * 4096]);
  float z = 0.0f;
  for (int o = 0; o < 512; ++o) z += __expf(sp[(size_t)o * 4096] - m);
  M[idx]  = m;
  Zr[idx] = 1.0f / z;
}

// ---------------------------------------------------------------------------
// Kernel 4: out[b,o,w,h] = sum_x softmax(s)[b,o,w,x] * v[b,o,x,h]
//   attn tile built in registers (exp(s-m)*Zr) -> bf16 LDS [w][x] (A, row reads).
//   v tile staged TRANSPOSED [h][x] so B-frag K(x) reads are contiguous.
// ---------------------------------------------------------------------------
__global__ __launch_bounds__(128) void sab_attn_out_wmma(
    const float* __restrict__ S, const __bf16* __restrict__ V,
    const float* __restrict__ M, const float* __restrict__ Zr,
    float* __restrict__ Out)
{
  constexpr int P = 4096, O = 512;
  const int o = blockIdx.x, b = blockIdx.y;
  const int t = threadIdx.x, wid = t >> 5, lane = t & 31;
  const int mrow = lane & 15;
  const int kA = (lane < 16) ? 0 : 16;
  const int kB = (lane < 16) ? 0 : 16;

  __shared__ __bf16 at[64][68];   // attn [w][x]
  __shared__ __bf16 vtT[64][68];  // v transposed: [h][x]

  const size_t base = ((size_t)b * O + o) * P;
  const size_t mz   = (size_t)b * 4096;
  // attn tile: row-major float4 reads of S/M/Zr, exp+normalize, bf16 to LDS
#pragma unroll
  for (int i = 0; i < 8; ++i) {
    const int flat = (t + i * 128) * 4;
    const int r = flat >> 6, c = flat & 63;
    const float4 s4 = *reinterpret_cast<const float4*>(&S[base + flat]);
    const float4 m4 = *reinterpret_cast<const float4*>(&M[mz + flat]);
    const float4 z4 = *reinterpret_cast<const float4*>(&Zr[mz + flat]);
    v4bf av;
    av[0] = (__bf16)(__expf(s4.x - m4.x) * z4.x);
    av[1] = (__bf16)(__expf(s4.y - m4.y) * z4.y);
    av[2] = (__bf16)(__expf(s4.z - m4.z) * z4.z);
    av[3] = (__bf16)(__expf(s4.w - m4.w) * z4.w);
    *reinterpret_cast<v4bf*>(&at[r][c]) = av;
  }
  // v tile transposed: vtT[h][x] = V[b,o,x,h]; adjacent lanes read adjacent h.
#pragma unroll
  for (int i = 0; i < 8; ++i) {
    const int chunk = t + i * 128;   // 1024 chunks of 4 elems
    const int h = chunk & 63;
    const int x = (chunk >> 6) * 4;
    const __bf16* __restrict__ vp = &V[base + (size_t)x * 64 + h];
    v4bf hv;
    hv[0] = vp[0];
    hv[1] = vp[64];
    hv[2] = vp[128];
    hv[3] = vp[192];
    *reinterpret_cast<v4bf*>(&vtT[h][x]) = hv;
  }
  __syncthreads();

  v8f acc0 = {}, acc1 = {}, acc2 = {}, acc3 = {};
  const int w0 = wid * 16;
#pragma unroll
  for (int kk = 0; kk < 64; kk += 32) {  // K over x
    v16bf a;
#pragma unroll
    for (int i = 0; i < 8; ++i) {
      a[i]     = at[w0 + mrow][kk + kA + i];
      a[i + 8] = at[w0 + mrow][kk + kA + 8 + i];
    }
    v16bf b0, b1, b2, b3;  // B[k=x][n=h] -> vtT[h][x]: contiguous per lane
#pragma unroll
    for (int i = 0; i < 16; ++i) {
      b0[i] = vtT[ 0 + mrow][kk + kB + i];
      b1[i] = vtT[16 + mrow][kk + kB + i];
      b2[i] = vtT[32 + mrow][kk + kB + i];
      b3[i] = vtT[48 + mrow][kk + kB + i];
    }
    acc0 = WMMA_BF16(a, b0, acc0);
    acc1 = WMMA_BF16(a, b1, acc1);
    acc2 = WMMA_BF16(a, b2, acc2);
    acc3 = WMMA_BF16(a, b3, acc3);
  }

  const int rb = (lane < 16) ? 0 : 8;
#pragma unroll
  for (int r = 0; r < 8; ++r) {
    float* op = &Out[base + (size_t)(w0 + rb + r) * 64 + mrow];
    op[ 0] = acc0[r];
    op[16] = acc1[r];
    op[32] = acc2[r];
    op[48] = acc3[r];
  }
}

// ---------------------------------------------------------------------------
extern "C" void kernel_launch(void* const* d_in, const int* in_sizes, int n_in,
                              void* d_out, int out_size, void* d_ws, size_t ws_size,
                              hipStream_t stream) {
  (void)in_sizes; (void)n_in; (void)out_size; (void)ws_size;
  const float* q  = (const float*)d_in[0];
  const float* k  = (const float*)d_in[1];
  const float* v  = (const float*)d_in[2];
  const float* wq = (const float*)d_in[3];
  const float* bq = (const float*)d_in[4];
  const float* wk = (const float*)d_in[5];
  const float* bk = (const float*)d_in[6];
  const float* wv = (const float*)d_in[7];
  const float* bv = (const float*)d_in[8];

  // Workspace layout: qb/kb/vb (bf16, 67MB each), S (fp32, 134MB), M/Zr (256KB each)
  const size_t NE = (size_t)16 * 512 * 4096;
  __bf16* qb = (__bf16*)d_ws;
  __bf16* kb = qb + NE;
  __bf16* vb = kb + NE;
  float*  S  = (float*)(vb + NE);
  float*  M  = S + NE;
  float*  Zr = M + (size_t)16 * 4096;

  dim3 gp(64, 8, 16);   // (P/64, O/64, B)
  sab_proj_bf16_wmma<<<gp, 128, 0, stream>>>(q, wq, bq, qb);
  sab_proj_bf16_wmma<<<gp, 128, 0, stream>>>(k, wk, bk, kb);
  sab_proj_bf16_wmma<<<gp, 128, 0, stream>>>(v, wv, bv, vb);

  dim3 ga(512, 16);     // (O, B)
  const float rscale = 0.044194173824159216f;  // 1/sqrt(512)
  sab_scores_wmma<<<ga, 128, 0, stream>>>(qb, kb, S, rscale);
  sab_softmax_stats<<<256, 256, 0, stream>>>(S, M, Zr);
  sab_attn_out_wmma<<<ga, 128, 0, stream>>>(S, vb, M, Zr, (float*)d_out);
}
